// GINSWDSingleModel_10831907520997
// MI455X (gfx1250) — matrix operand order repaired and checked by reading, not compiled, not run on hardware
//
#include <hip/hip_runtime.h>

#define N_NODES 32768
#define NPG     256
#define BSZ     128
#define NE      524288
#define DIM     64
#define NC      16
#define KAT     256
#define BN_EPS  1e-5f
#define GAMMA   (1.0f / 64.0f)
#define SLOPE   0.1f

typedef __attribute__((ext_vector_type(16))) __bf16 v16bf;
typedef __attribute__((ext_vector_type(8)))  __bf16 v8bf;
typedef __attribute__((ext_vector_type(8)))  float  v8f;

// ---- async global->LDS staging (gfx1250 GLOBAL_LOAD_ASYNC_TO_LDS, ASYNCcnt) ----
// All 256 threads participate; each lane moves 16B per issue (4 KB per block-iteration).
__device__ __forceinline__ void async_copy_to_lds(void* lds_dst, const void* gsrc, int bytes) {
    unsigned l = (unsigned)(uintptr_t)lds_dst + threadIdx.x * 16u;   // addr[31:0] = LDS offset
    const char* g = (const char*)gsrc + threadIdx.x * 16u;
    for (int off = 0; off < bytes; off += 4096) {
        asm volatile("global_load_async_to_lds_b128 %0, %1, off"
                     :: "v"(l + (unsigned)off), "v"((const void*)(g + off))
                     : "memory");
    }
}
__device__ __forceinline__ void wait_async() {
    asm volatile("s_wait_asynccnt 0x0" ::: "memory");
}

// ---- assemble a v16bf WMMA operand from two 16-byte-aligned chunks ----
__device__ __forceinline__ v16bf ld16(const __bf16* p0, const __bf16* p1) {
    union { v16bf v; v8bf h[2]; } u;
    u.h[0] = *(const v8bf*)p0;
    u.h[1] = *(const v8bf*)p1;
    return u.v;
}

// A (16x32, MxK) from bf16 row-major [rows x 64]
__device__ __forceinline__ v16bf load_a_bf(const __bf16* M, int m0, int k0, int lane) {
    const __bf16* base = M + (m0 + (lane & 15)) * DIM + k0 + 8 * ((lane >> 4) & 1);
    return ld16(base, base + 16);
}

// B (32x16, KxN) where B[k][j] = M[j][k], M bf16 row-major (S = A * M^T)
__device__ __forceinline__ v16bf load_bt_bf(const __bf16* M, int k0, int n0, int lane) {
    const __bf16* base = M + (n0 + (lane & 15)) * DIM + k0 + 8 * ((lane >> 4) & 1);
    return ld16(base, base + 16);
}

// A (16x32) from f32 row-major with on-the-fly bf16 convert
__device__ __forceinline__ v16bf load_a_f32(const float* M, int m0, int k0, int lane) {
    const float* p = M + (m0 + (lane & 15)) * DIM + k0 + 8 * ((lane >> 4) & 1);
    v16bf a;
#pragma unroll
    for (int v = 0; v < 8; ++v) {
        int kb = ((v >= 4) ? 16 : 0) + 2 * (v & 3);
        a[2 * v]     = (__bf16)p[kb];
        a[2 * v + 1] = (__bf16)p[kb + 1];
    }
    return a;
}

#define WMMA_BF16(A, B, C) \
    __builtin_amdgcn_wmma_f32_16x16x32_bf16(false, (A), false, (B), (short)0, (C), false, false)

// -------------------- prep kernels --------------------
__global__ void f32_to_bf16_k(const float2* __restrict__ src, __bf16* __restrict__ dst, int n2) {
    int i = blockIdx.x * blockDim.x + threadIdx.x;
    if (i < n2) {
        float2 f = src[i];
        dst[2 * i]     = (__bf16)f.x;
        dst[2 * i + 1] = (__bf16)f.y;
    }
}

// Convert + transpose 9 weight matrices (64x64): wt[slot][col*64+k] = W[k*64+col], slot = w*3+l
__global__ void prep_wt(const float* __restrict__ W1, const float* __restrict__ W2,
                        const float* __restrict__ W3, __bf16* __restrict__ wt) {
    int slot = blockIdx.x;            // 0..8
    int w = slot / 3, l = slot % 3;
    const float* src = (w == 0 ? W1 : (w == 1 ? W2 : W3)) + l * DIM * DIM;
    __bf16* dst = wt + slot * DIM * DIM;
    for (int e = threadIdx.x; e < DIM * DIM; e += blockDim.x) {
        int c = e >> 6, k = e & 63;
        dst[e] = (__bf16)src[k * DIM + c];
    }
}

// -------------------- elementwise / utility kernels --------------------
__global__ void copy_f4(const float4* __restrict__ src, float4* __restrict__ dst, int n) {
    int i = blockIdx.x * blockDim.x + threadIdx.x;
    if (i < n) dst[i] = src[i];
}

// z[dst] += h[src]  (z pre-initialized to h, giving z = h + aggregate)
__global__ void scatter_add(const float* __restrict__ h, const int* __restrict__ esrc,
                            const int* __restrict__ edst, float* __restrict__ z) {
    int t = blockIdx.x * blockDim.x + threadIdx.x;
    int e = t >> 4;
    if (e >= NE) return;
    int c4 = (t & 15) * 4;
    int s = esrc[e], d = edst[e];
    const float* hp = h + s * DIM + c4;
    float* zp = z + d * DIM + c4;
    atomicAdd(zp + 0, hp[0]);
    atomicAdd(zp + 1, hp[1]);
    atomicAdd(zp + 2, hp[2]);
    atomicAdd(zp + 3, hp[3]);
}

// out = Z[N x 64] @ W[64 x 64] + bias, W given as bf16 transposed (wt[col*64+k]).
// 256 blocks x 8 waves; each wave: one 16-row stripe, all four 16-col tiles (8 WMMAs).
__global__ void gemm_bias(const float* __restrict__ Z, const __bf16* __restrict__ WT,
                          const float* __restrict__ bias, float* __restrict__ out) {
    __shared__ __bf16 s_wt[DIM * DIM];     // 8 KB
    async_copy_to_lds(s_wt, WT, DIM * DIM * 2);
    wait_async();
    __syncthreads();

    int lane = threadIdx.x & 31;
    int wv   = threadIdx.x >> 5;
    int m0   = (blockIdx.x * 8 + wv) * 16;
    v8f acc[4] = {};
#pragma unroll
    for (int k0 = 0; k0 < DIM; k0 += 32) {
        v16bf a = load_a_f32(Z, m0, k0, lane);
#pragma unroll
        for (int n = 0; n < 4; ++n) {
            v16bf b = load_bt_bf(s_wt, k0, n * 16, lane);
            acc[n] = WMMA_BF16(a, b, acc[n]);
        }
    }
    int g = (lane >> 4) & 1;
    int cl = lane & 15;
#pragma unroll
    for (int n = 0; n < 4; ++n) {
        int col = n * 16 + cl;
        float bb = bias[col];
#pragma unroll
        for (int v = 0; v < 8; ++v) {
            int row = m0 + v + 8 * g;
            out[row * DIM + col] = acc[n][v] + bb;
        }
    }
}

__global__ void zero_stats(float* stats) {
    stats[threadIdx.x] = 0.0f;
}

__global__ void col_stats(const float* __restrict__ X, float* __restrict__ stats) {
    int col = threadIdx.x & 63;
    int r0  = blockIdx.x * 4 + (threadIdx.x >> 6);
    float s1 = 0.0f, s2 = 0.0f;
    for (int row = r0; row < N_NODES; row += gridDim.x * 4) {
        float x = X[row * DIM + col];
        s1 += x;
        s2 += x * x;
    }
    atomicAdd(&stats[col], s1);
    atomicAdd(&stats[64 + col], s2);
}

__global__ void bn_lrelu(float* __restrict__ X, const float* __restrict__ stats,
                         const float* __restrict__ gm, const float* __restrict__ bt) {
    int i = blockIdx.x * blockDim.x + threadIdx.x;
    int col = i & 63;
    float m = stats[col] * (1.0f / N_NODES);
    float v = stats[64 + col] * (1.0f / N_NODES) - m * m;
    float y = gm[col] * (X[i] - m) * rsqrtf(v + BN_EPS) + bt[col];
    X[i] = (y >= 0.0f) ? y : SLOPE * y;
}

__global__ void row_sqnorm(const float* __restrict__ X, float* __restrict__ out, int rows) {
    int r = blockIdx.x * blockDim.x + threadIdx.x;
    if (r >= rows) return;
    const float* p = X + r * DIM;
    float s = 0.0f;
#pragma unroll
    for (int d = 0; d < DIM; ++d) s += p[d] * p[d];
    out[r] = s;
}

// -------------------- MMD kernels --------------------
// 32x32 super-tile body: 4 accumulators, 8 WMMAs per k-sweep, operand reuse 2x.
// Returns the lane's partial sum of exp(-gamma * dist).
__device__ __forceinline__ float mmd_supertile(const __bf16* s_rows, const __bf16* s_cols,
                                               const float* s_rn, const float* s_cn,
                                               int sm0, int sn0, int lane) {
    int g = (lane >> 4) & 1, cl = lane & 15;
    v8f a00 = {}, a01 = {}, a10 = {}, a11 = {};
#pragma unroll
    for (int k0 = 0; k0 < DIM; k0 += 32) {
        v16bf fa0 = load_a_bf(s_rows, sm0, k0, lane);
        v16bf fa1 = load_a_bf(s_rows, sm0 + 16, k0, lane);
        v16bf fb0 = load_bt_bf(s_cols, k0, sn0, lane);
        v16bf fb1 = load_bt_bf(s_cols, k0, sn0 + 16, lane);
        a00 = WMMA_BF16(fa0, fb0, a00);
        a01 = WMMA_BF16(fa0, fb1, a01);
        a10 = WMMA_BF16(fa1, fb0, a10);
        a11 = WMMA_BF16(fa1, fb1, a11);
    }
    float c0 = s_cn[sn0 + cl];
    float c1 = s_cn[sn0 + 16 + cl];
    float acc = 0.0f;
#pragma unroll
    for (int v = 0; v < 8; ++v) {
        float r0 = s_rn[sm0 + v + 8 * g];
        float r1 = s_rn[sm0 + 16 + v + 8 * g];
        acc += __expf(-GAMMA * (r0 + c0 - 2.0f * a00[v]));
        acc += __expf(-GAMMA * (r0 + c1 - 2.0f * a01[v]));
        acc += __expf(-GAMMA * (r1 + c0 - 2.0f * a10[v]));
        acc += __expf(-GAMMA * (r1 + c1 - 2.0f * a11[v]));
    }
    return acc;
}

// Self Gram: one block per 256-row slice; out[b] = mean exp(-gamma * ||xi - xj||^2)
__global__ void mmd_self(const __bf16* __restrict__ M, const float* __restrict__ norms,
                         float* __restrict__ out) {
    __shared__ __bf16 s_m[KAT * DIM];   // 32 KB
    __shared__ float  s_n[256];
    __shared__ float  red[256];
    int bidx = blockIdx.x;
    async_copy_to_lds(s_m, M + bidx * KAT * DIM, KAT * DIM * 2);
    s_n[threadIdx.x] = norms[bidx * KAT + threadIdx.x];
    wait_async();
    __syncthreads();

    int lane = threadIdx.x & 31, wv = threadIdx.x >> 5;
    float acc = 0.0f;
    for (int st = wv; st < 64; st += 8) {             // 8x8 grid of 32x32 super-tiles
        int sm0 = (st >> 3) << 5, sn0 = (st & 7) << 5;
        acc += mmd_supertile(s_m, s_m, s_n, s_n, sm0, sn0, lane);
    }
    red[threadIdx.x] = acc;
    __syncthreads();
    for (int s = 128; s > 0; s >>= 1) {
        if ((int)threadIdx.x < s) red[threadIdx.x] += red[threadIdx.x + s];
        __syncthreads();
    }
    if (threadIdx.x == 0) out[bidx] = red[0] * (1.0f / 65536.0f);
}

// Cross term + final combine: out[b*16+c] = kxx[b] + kyy[c] - 2*mean exp(-g*||x-a||^2)
__global__ void mmd_xy(const __bf16* __restrict__ X, const __bf16* __restrict__ atoms,
                       const float* __restrict__ xn, const float* __restrict__ an,
                       const float* __restrict__ kxx, const float* __restrict__ kyy,
                       float* __restrict__ out) {
    __shared__ __bf16 s_x[NPG * DIM];   // 32 KB
    __shared__ __bf16 s_a[KAT * DIM];   // 32 KB
    __shared__ float  s_xn[256];
    __shared__ float  s_an[256];
    __shared__ float  red[256];
    int b = blockIdx.x >> 4, c = blockIdx.x & 15;
    async_copy_to_lds(s_x, X + b * NPG * DIM, NPG * DIM * 2);
    async_copy_to_lds(s_a, atoms + c * KAT * DIM, KAT * DIM * 2);
    s_xn[threadIdx.x] = xn[b * NPG + threadIdx.x];
    s_an[threadIdx.x] = an[c * KAT + threadIdx.x];
    wait_async();
    __syncthreads();

    int lane = threadIdx.x & 31, wv = threadIdx.x >> 5;
    float acc = 0.0f;
    for (int st = wv; st < 64; st += 8) {             // 8x8 grid of 32x32 super-tiles
        int sm0 = (st >> 3) << 5, sn0 = (st & 7) << 5;
        acc += mmd_supertile(s_x, s_a, s_xn, s_an, sm0, sn0, lane);
    }
    red[threadIdx.x] = acc;
    __syncthreads();
    for (int s = 128; s > 0; s >>= 1) {
        if ((int)threadIdx.x < s) red[threadIdx.x] += red[threadIdx.x + s];
        __syncthreads();
    }
    if (threadIdx.x == 0)
        out[b * NC + c] = kxx[b] + kyy[c] - 2.0f * (red[0] * (1.0f / 65536.0f));
}

// -------------------- host launcher --------------------
extern "C" void kernel_launch(void* const* d_in, const int* in_sizes, int n_in,
                              void* d_out, int out_size, void* d_ws, size_t ws_size,
                              hipStream_t stream) {
    const float* x     = (const float*)d_in[0];
    const float* W1    = (const float*)d_in[1];
    const float* b1    = (const float*)d_in[2];
    const float* W2    = (const float*)d_in[3];
    const float* b2    = (const float*)d_in[4];
    const float* W3    = (const float*)d_in[5];
    const float* b3    = (const float*)d_in[6];
    const float* g1    = (const float*)d_in[7];
    const float* be1   = (const float*)d_in[8];
    const float* g2    = (const float*)d_in[9];
    const float* be2   = (const float*)d_in[10];
    const float* atoms = (const float*)d_in[11];
    const int*   ei    = (const int*)d_in[12];
    const int* esrc = ei;
    const int* edst = ei + NE;

    float* ws    = (float*)d_ws;
    float* bufA  = ws;                        // N*D f32
    float* bufB  = bufA + N_NODES * DIM;      // N*D f32
    float* stats = bufB + N_NODES * DIM;      // 128 f32
    float* xn    = stats + 128;               // N f32
    float* an    = xn + N_NODES;              // C*K f32
    float* kxx   = an + NC * KAT;             // 128 f32
    float* kyy   = kxx + BSZ;                 // 16 f32
    __bf16* hbf     = (__bf16*)(kyy + 16);            // N*D bf16
    __bf16* atomsbf = hbf + N_NODES * DIM;            // C*K*D bf16
    __bf16* wt      = atomsbf + NC * KAT * DIM;       // 9*64*64 bf16
    float* out   = (float*)d_out;

    const int TB = 256;
    const int ND4 = N_NODES * DIM / 4;

    // weight convert+transpose (deterministic, re-run every call)
    prep_wt<<<9, TB, 0, stream>>>(W1, W2, W3, wt);

    copy_f4<<<ND4 / TB, TB, 0, stream>>>((const float4*)x, (float4*)bufA, ND4);

    for (int l = 0; l < 3; ++l) {
        // z = h + segment_sum(h[src], dst)
        copy_f4<<<ND4 / TB, TB, 0, stream>>>((const float4*)bufA, (float4*)bufB, ND4);
        scatter_add<<<(NE * 16) / TB, TB, 0, stream>>>(bufA, esrc, edst, bufB);
        // mlp layer 1
        gemm_bias<<<256, TB, 0, stream>>>(bufB, wt + (0 * 3 + l) * DIM * DIM, b1 + l * DIM, bufA);
        zero_stats<<<1, 128, 0, stream>>>(stats);
        col_stats<<<128, TB, 0, stream>>>(bufA, stats);
        bn_lrelu<<<N_NODES * DIM / TB, TB, 0, stream>>>(bufA, stats, g1 + l * DIM, be1 + l * DIM);
        // mlp layer 2
        gemm_bias<<<256, TB, 0, stream>>>(bufA, wt + (1 * 3 + l) * DIM * DIM, b2 + l * DIM, bufB);
        zero_stats<<<1, 128, 0, stream>>>(stats);
        col_stats<<<128, TB, 0, stream>>>(bufB, stats);
        bn_lrelu<<<N_NODES * DIM / TB, TB, 0, stream>>>(bufB, stats, g2 + l * DIM, be2 + l * DIM);
        // mlp layer 3 (linear)
        gemm_bias<<<256, TB, 0, stream>>>(bufB, wt + (2 * 3 + l) * DIM * DIM, b3 + l * DIM, bufA);
    }

    // norms in f32 (exact), matrices converted to bf16 once for WMMA feeding
    row_sqnorm<<<N_NODES / TB, TB, 0, stream>>>(bufA, xn, N_NODES);
    row_sqnorm<<<(NC * KAT) / TB, TB, 0, stream>>>(atoms, an, NC * KAT);
    f32_to_bf16_k<<<(N_NODES * DIM / 2) / TB, TB, 0, stream>>>((const float2*)bufA, hbf, N_NODES * DIM / 2);
    f32_to_bf16_k<<<(NC * KAT * DIM / 2) / TB, TB, 0, stream>>>((const float2*)atoms, atomsbf, NC * KAT * DIM / 2);

    mmd_self<<<NC, TB, 0, stream>>>(atomsbf, an, kyy);
    mmd_self<<<BSZ, TB, 0, stream>>>(hbf, xn, kxx);
    mmd_xy<<<BSZ * NC, TB, 0, stream>>>(hbf, atomsbf, xn, an, kxx, kyy, out);
}
